// EncoderLayer_79766132622004
// MI455X (gfx1250) — matrix operand order repaired and checked by reading, not compiled, not run on hardware
//
#include <hip/hip_runtime.h>

// ---------------------------------------------------------------------------
// EncoderLayer for MI455X (gfx1250, wave32, WMMA).
// All dense GEMMs use v_wmma_f32_16x16x32_f16 (f16 in, fp32 accumulate) with
// double-buffered LDS tiles; fragments are contiguous 16B chunks so every
// fragment build is two ds_load_b128. Input BN-apply/leaky is fused into the
// GEMM staging path (templated, branch-free, vectorized scale loads).
// ---------------------------------------------------------------------------

#define HH   192
#define WW   192
#define CC   256
#define NN   36864        // HH*WW
#define EE   294912
#define KK   289
#define SLOPE 0.01f
#define EPSV  1e-5f

typedef _Float16 h16v __attribute__((ext_vector_type(16)));
typedef _Float16 h8v  __attribute__((ext_vector_type(8)));
typedef _Float16 h4v  __attribute__((ext_vector_type(4)));
typedef float    f8v  __attribute__((ext_vector_type(8)));

__device__ __forceinline__ float lrelu(float x) { return (x >= 0.f) ? x : SLOPE * x; }

// ---------------------------------------------------------------------------
// WMMA GEMM:  Out (Co x NN, planar) = Wm (Co x Ci) @ xform(In (Ci x NN))
//   AFF : per-channel affine (fused BN apply) on input
//   LK  : leaky on input;  BIAS/OLK : per-row bias / leaky on output
// Block: 256 threads = 8 waves (2x4), block tile 32(M) x 64(N), K step 32.
// A tile [m][k], B tile [n][k] (transposed), 40-half pitch, double-buffered.
// ---------------------------------------------------------------------------
template<int AFF, int LK, int BIAS, int OLK>
__global__ __launch_bounds__(256)
void gemm_wmma_k(const float* __restrict__ Wm, const float* __restrict__ In,
                 float* __restrict__ Out, int Co, int Ci,
                 const float* __restrict__ isc, const float* __restrict__ ish,
                 const float* __restrict__ bias)
{
    __shared__ alignas(16) _Float16 As[2][32][40];   // [buf][m][k], pitch 80B
    __shared__ alignas(16) _Float16 Bs[2][64][40];   // [buf][n][k], pitch 80B
    const int tid  = threadIdx.x;
    const int lane = tid & 31;
    const int wv   = tid >> 5;
    const int wm   = wv >> 2;         // 0..1
    const int wn   = wv & 3;          // 0..3
    const int m0   = blockIdx.y * 32;
    const int n0   = blockIdx.x * 64;

    // A staging: thread -> (m = tid>>3, k0 = (tid&7)*4), one float4 load
    const int am  = tid >> 3;
    const int ak0 = (tid & 7) * 4;
    const int amg = m0 + am;
    const int amr = (amg < Co) ? amg : (Co - 1);
    const float amask = (amg < Co) ? 1.f : 0.f;
    const float* aptr = Wm + (size_t)amr * Ci + ak0;

    // B staging: thread -> (n = tid&63, k0 = (tid>>6)*8), 8 coalesced b32 loads
    const int bn  = tid & 63;
    const int bk0 = (tid >> 6) * 8;
    const float* bptr = In + (size_t)bk0 * NN + n0 + bn;

    auto stage = [&](int buf, int kt) {
        {   // A
            const float4 w4 = *(const float4*)(aptr + kt);
            h4v p;
            p[0] = (_Float16)(w4.x * amask);
            p[1] = (_Float16)(w4.y * amask);
            p[2] = (_Float16)(w4.z * amask);
            p[3] = (_Float16)(w4.w * amask);
            *(h4v*)(&As[buf][am][ak0]) = p;
        }
        {   // B
            const float* bp = bptr + (size_t)kt * NN;
            float s[8], t[8];
            if (AFF) {
                const int ci0 = kt + bk0;
                float4 s0 = *(const float4*)(isc + ci0);
                float4 s1 = *(const float4*)(isc + ci0 + 4);
                float4 t0 = *(const float4*)(ish + ci0);
                float4 t1 = *(const float4*)(ish + ci0 + 4);
                s[0]=s0.x; s[1]=s0.y; s[2]=s0.z; s[3]=s0.w;
                s[4]=s1.x; s[5]=s1.y; s[6]=s1.z; s[7]=s1.w;
                t[0]=t0.x; t[1]=t0.y; t[2]=t0.z; t[3]=t0.w;
                t[4]=t1.x; t[5]=t1.y; t[6]=t1.z; t[7]=t1.w;
            }
            h8v p;
            #pragma unroll
            for (int j = 0; j < 8; ++j) {
                float v = bp[(size_t)j * NN];
                if (AFF) v = v * s[j] + t[j];
                if (LK)  v = lrelu(v);
                p[j] = (_Float16)v;
            }
            *(h8v*)(&Bs[buf][bn][bk0]) = p;
        }
    };

    // fragment geometry (ISA 7.12.2 layouts)
    const int hiA = (lane >= 16) ? 8 : 0;    // halves
    const int hiB = (lane >= 16) ? 8 : 0;    // halves
    const int mr  = wm * 16 + (lane & 15);
    const int nc  = wn * 16 + (lane & 15);

    f8v acc = {};

    stage(0, 0);
    __syncthreads();

    for (int kt = 0; kt < Ci; kt += 32) {
        const int cur = (kt >> 5) & 1;
        if (kt + 32 < Ci) stage(cur ^ 1, kt + 32);   // prefetch next tile

        h8v a0 = *(const h8v*)(&As[cur][mr][hiA]);        // K = hiA..hiA+7
        h8v a1 = *(const h8v*)(&As[cur][mr][16 + hiA]);   // K = 16+hiA..
        h8v b0 = *(const h8v*)(&Bs[cur][nc][hiB]);        // K = hiB..hiB+7
        h8v b1 = *(const h8v*)(&Bs[cur][nc][16 + hiB]);   // K = 16+hiB..
        h16v af = __builtin_shufflevector(a0, a1, 0,1,2,3,4,5,6,7,
                                                  8,9,10,11,12,13,14,15);
        h16v bf = __builtin_shufflevector(b0, b1, 0,1,2,3,4,5,6,7,
                                                  8,9,10,11,12,13,14,15);

        acc = __builtin_amdgcn_wmma_f32_16x16x32_f16(
                  false, af, false, bf, (short)0, acc, false, false);
        __syncthreads();
    }

    const int nco = n0 + wn * 16 + (lane & 15);
    #pragma unroll
    for (int r = 0; r < 8; ++r) {
        int m = m0 + wm * 16 + r + ((lane >= 16) ? 8 : 0);
        if (m < Co) {
            float v = acc[r];
            if (BIAS) v += bias[m];
            if (OLK)  v = lrelu(v);
            Out[(size_t)m * NN + nco] = v;
        }
    }
}

// ---------------------------------------------------------------------------
// Depthwise 17x17 conv (pad 8) with leaky on input, planar in/out.
// ---------------------------------------------------------------------------
__global__ __launch_bounds__(256)
void dwconv17_k(const float* __restrict__ in, const float* __restrict__ w,
                float* __restrict__ out)
{
    __shared__ float tile[32][32];
    __shared__ float wt[289];
    const int ch  = blockIdx.z;
    const int tid = threadIdx.y * 16 + threadIdx.x;
    const int gy0 = blockIdx.y * 16 - 8;
    const int gx0 = blockIdx.x * 16 - 8;
    const float* ip = in + (size_t)ch * NN;

    for (int i = tid; i < 1024; i += 256) {
        int ty = i >> 5, tx = i & 31;
        int gy = gy0 + ty, gx = gx0 + tx;
        float v = 0.f;
        if (gy >= 0 && gy < HH && gx >= 0 && gx < WW) v = lrelu(ip[gy * WW + gx]);
        tile[ty][tx] = v;
    }
    for (int i = tid; i < 289; i += 256) wt[i] = w[(size_t)ch * 289 + i];
    __syncthreads();

    float acc = 0.f;
    #pragma unroll 1
    for (int ky = 0; ky < 17; ++ky) {
        #pragma unroll
        for (int kx = 0; kx < 17; ++kx)
            acc += tile[threadIdx.y + ky][threadIdx.x + kx] * wt[ky * 17 + kx];
    }
    out[(size_t)ch * NN + (blockIdx.y * 16 + threadIdx.y) * WW
        + blockIdx.x * 16 + threadIdx.x] = acc;
}

// ---------------------------------------------------------------------------
// Depthwise-expand 3x3 conv (groups=C, 4 outputs per group), leaky on input.
// ---------------------------------------------------------------------------
__global__ __launch_bounds__(256)
void dwconv3x3_k(const float* __restrict__ in, const float* __restrict__ w,
                 float* __restrict__ out)
{
    __shared__ float tile[18][18];
    __shared__ float wt[9];
    const int oc  = blockIdx.z;
    const int ic  = oc >> 2;
    const int tid = threadIdx.y * 16 + threadIdx.x;
    const int gy0 = blockIdx.y * 16 - 1;
    const int gx0 = blockIdx.x * 16 - 1;
    const float* ip = in + (size_t)ic * NN;

    for (int i = tid; i < 324; i += 256) {
        int ty = i / 18, tx = i % 18;
        int gy = gy0 + ty, gx = gx0 + tx;
        float v = 0.f;
        if (gy >= 0 && gy < HH && gx >= 0 && gx < WW) v = lrelu(ip[gy * WW + gx]);
        tile[ty][tx] = v;
    }
    if (tid < 9) wt[tid] = w[(size_t)oc * 9 + tid];
    __syncthreads();

    float acc = 0.f;
    #pragma unroll
    for (int ky = 0; ky < 3; ++ky)
        #pragma unroll
        for (int kx = 0; kx < 3; ++kx)
            acc += tile[threadIdx.y + ky][threadIdx.x + kx] * wt[ky * 3 + kx];

    out[(size_t)oc * NN + (blockIdx.y * 16 + threadIdx.y) * WW
        + blockIdx.x * 16 + threadIdx.x] = acc;
}

// ---------------------------------------------------------------------------
// Per-channel BN stats over planar (Cch x NN) -> scale/shift (folded g,b).
// ---------------------------------------------------------------------------
__global__ __launch_bounds__(256)
void stats_planar_k(const float* __restrict__ buf, int Nn,
                    const float* __restrict__ g, const float* __restrict__ b,
                    float* __restrict__ sc, float* __restrict__ sh)
{
    __shared__ float r1[256], r2[256];
    const int c = blockIdx.x, tid = threadIdx.x;
    float a = 0.f, q = 0.f;
    const float* p = buf + (size_t)c * Nn;
    for (int n = tid; n < Nn; n += 256) { float v = p[n]; a += v; q += v * v; }
    r1[tid] = a; r2[tid] = q; __syncthreads();
    for (int s = 128; s > 0; s >>= 1) {
        if (tid < s) { r1[tid] += r1[tid + s]; r2[tid] += r2[tid + s]; }
        __syncthreads();
    }
    if (tid == 0) {
        float mean = r1[0] / (float)Nn;
        float var  = r2[0] / (float)Nn - mean * mean;
        float s    = g[c] * rsqrtf(var + EPSV);
        sc[c] = s; sh[c] = b[c] - mean * s;
    }
}

// BN1d stats of X_trans = XA/(A_sum+eps), row-major (NN x CC), per channel c.
__global__ __launch_bounds__(256)
void stats_xtrans_k(const float* __restrict__ xa, const float* __restrict__ asum,
                    const float* __restrict__ g, const float* __restrict__ b,
                    float* __restrict__ sc, float* __restrict__ sh)
{
    __shared__ float r1[256], r2[256];
    const int c = blockIdx.x, tid = threadIdx.x;
    float a = 0.f, q = 0.f;
    for (int n = tid; n < NN; n += 256) {
        float v = xa[(size_t)n * CC + c] / (asum[n] + EPSV);
        a += v; q += v * v;
    }
    r1[tid] = a; r2[tid] = q; __syncthreads();
    for (int s = 128; s > 0; s >>= 1) {
        if (tid < s) { r1[tid] += r1[tid + s]; r2[tid] += r2[tid + s]; }
        __syncthreads();
    }
    if (tid == 0) {
        float mean = r1[0] / (float)NN;
        float var  = r2[0] / (float)NN - mean * mean;
        float s    = g[c] * rsqrtf(var + EPSV);
        sc[c] = s; sh[c] = b[c] - mean * s;
    }
}

// img += t*sc[c] + sh[c]  (planar C x NN)
__global__ __launch_bounds__(256)
void add_bn_k(float* __restrict__ img, const float* __restrict__ t,
              const float* __restrict__ sc, const float* __restrict__ sh)
{
    size_t total = (size_t)CC * NN;
    for (size_t i = (size_t)blockIdx.x * 256 + threadIdx.x; i < total;
         i += (size_t)gridDim.x * 256) {
        int c = (int)(i / NN);
        img[i] += t[i] * sc[c] + sh[c];
    }
}

// out_row = xn_row + sc[c]*(XA/(A+eps)) + sh[c]   (row major N x C)
__global__ __launch_bounds__(256)
void xtrans_apply_k(const float* __restrict__ xn, const float* __restrict__ xa,
                    const float* __restrict__ asum, const float* __restrict__ sc,
                    const float* __restrict__ sh, float* __restrict__ out)
{
    size_t total = (size_t)CC * NN;
    for (size_t i = (size_t)blockIdx.x * 256 + threadIdx.x; i < total;
         i += (size_t)gridDim.x * 256) {
        int c = (int)(i % CC);
        int n = (int)(i / CC);
        float v = xa[i] / (asum[n] + EPSV);
        out[i] = xn[i] + sc[c] * v + sh[c];
    }
}

__global__ __launch_bounds__(256)
void zero_k(float* __restrict__ p, size_t n)
{
    for (size_t i = (size_t)blockIdx.x * 256 + threadIdx.x; i < n;
         i += (size_t)gridDim.x * 256) p[i] = 0.f;
}

// Tiled transpose: dst[j*R + i] = src[i*Cc + j]   (src R x Cc row-major)
__global__ __launch_bounds__(256)
void transpose_k(const float* __restrict__ src, float* __restrict__ dst,
                 int R, int Cc)
{
    __shared__ float tile[32][33];
    int bx = blockIdx.x * 32, by = blockIdx.y * 32;
    #pragma unroll
    for (int j = 0; j < 4; ++j) {
        int r = by + threadIdx.y + j * 8, cx = bx + threadIdx.x;
        if (r < R && cx < Cc)
            tile[threadIdx.y + j * 8][threadIdx.x] = src[(size_t)r * Cc + cx];
    }
    __syncthreads();
    #pragma unroll
    for (int j = 0; j < 4; ++j) {
        int cy = bx + threadIdx.y + j * 8, rx = by + threadIdx.x;
        if (cy < Cc && rx < R)
            dst[(size_t)cy * R + rx] = tile[threadIdx.x][threadIdx.y + j * 8];
    }
}

// ---------------------------------------------------------------------------
// Edge scatter: one wave (32 lanes) per edge, 8 channels per lane, coalesced
// row-major gather/scatter + float atomics.
// ---------------------------------------------------------------------------
__global__ __launch_bounds__(256)
void edge_k(const int* __restrict__ info, const float* __restrict__ msk,
            const float* __restrict__ sig, const float* __restrict__ xn,
            float* __restrict__ xa, float* __restrict__ asum)
{
    const int lane = threadIdx.x & 31;
    const int wv   = threadIdx.x >> 5;
    const int e    = blockIdx.x * 8 + wv;
    if (e >= EE) return;
    const int* ip = info + (size_t)e * 4;
    int src = ip[0], k1 = ip[1], dst = ip[2], k2 = ip[3];
    float s = sig[(size_t)k1 * NN + src] + sig[(size_t)k2 * NN + dst];
    s = fminf(5.f, fmaxf(-5.f, s));
    float a = expf(s) * msk[e];
    if (lane == 0) atomicAdd(asum + dst, a);
    const float* xr = xn + (size_t)src * CC;
    float*       xw = xa + (size_t)dst * CC;
    #pragma unroll
    for (int j = 0; j < 8; ++j) {
        int c = lane + j * 32;
        atomicAdd(xw + c, xr[c] * a);
    }
}

// Final: d_out[n*C+c] = t[c][n]*sc[c]+sh[c] + img[c][n]   (tiled transpose)
__global__ __launch_bounds__(256)
void final_out_k(const float* __restrict__ t, const float* __restrict__ img,
                 const float* __restrict__ sc, const float* __restrict__ sh,
                 float* __restrict__ out)
{
    __shared__ float tile[32][33];
    int c0 = blockIdx.y * 32, n0 = blockIdx.x * 32;
    #pragma unroll
    for (int j = 0; j < 4; ++j) {
        int c = c0 + threadIdx.y + j * 8;
        int n = n0 + threadIdx.x;
        tile[threadIdx.y + j * 8][threadIdx.x] =
            t[(size_t)c * NN + n] * sc[c] + sh[c] + img[(size_t)c * NN + n];
    }
    __syncthreads();
    #pragma unroll
    for (int j = 0; j < 4; ++j) {
        int n = n0 + threadIdx.y + j * 8;
        out[(size_t)n * CC + c0 + threadIdx.x] = tile[threadIdx.x][threadIdx.y + j * 8];
    }
}

// ---------------------------------------------------------------------------
extern "C" void kernel_launch(void* const* d_in, const int* in_sizes, int n_in,
                              void* d_out, int out_size, void* d_ws, size_t ws_size,
                              hipStream_t stream)
{
    const float* X       = (const float*)d_in[0];
    const int*   InfoIdx = (const int*)  d_in[1];
    const float* edgmsk  = (const float*)d_in[2];
    const float* ppm_w0  = (const float*)d_in[3];
    const float* ppm_g0  = (const float*)d_in[4];
    const float* ppm_b0  = (const float*)d_in[5];
    const float* ppm_w1  = (const float*)d_in[6];
    const float* ppm_g1  = (const float*)d_in[7];
    const float* ppm_b1  = (const float*)d_in[8];
    const float* ffn1_w0 = (const float*)d_in[9];
    const float* ffn1_g0 = (const float*)d_in[10];
    const float* ffn1_b0 = (const float*)d_in[11];
    const float* ffn1_w1 = (const float*)d_in[12];
    const float* ffn1_g1 = (const float*)d_in[13];
    const float* ffn1_b1 = (const float*)d_in[14];
    const float* dis_w1  = (const float*)d_in[15];
    const float* dis_b1  = (const float*)d_in[16];
    const float* dis_w2  = (const float*)d_in[17];
    const float* dis_b2  = (const float*)d_in[18];
    const float* bn_g    = (const float*)d_in[19];
    const float* bn_b    = (const float*)d_in[20];
    const float* ffn2_w0 = (const float*)d_in[21];
    const float* ffn2_g0 = (const float*)d_in[22];
    const float* ffn2_b0 = (const float*)d_in[23];
    const float* ffn2_w1 = (const float*)d_in[24];
    const float* ffn2_g1 = (const float*)d_in[25];
    const float* ffn2_b1 = (const float*)d_in[26];

    float* out = (float*)d_out;
    float* ws  = (float*)d_ws;

    const size_t CN   = (size_t)CC * NN;     // 9,437,184 floats
    const size_t SIGF = (size_t)KK * NN;     // 10,653,696 floats

    float* B_img = ws;                 // planar C x N (img accumulator)
    float* B_t   = ws + CN;            // temp A  -- also start of E4 region
    float* B_u   = ws + 2 * CN;        // temp B
    float* B_h   = ws + 3 * CN;        // h (dis hidden)
    float* B_sig = ws + 4 * CN;        // signal planar (K x N)
    float* R_xn  = ws + 4 * CN + SIGF; // row-major Xn (N x C)
    float* XA    = R_xn + CN;          // row-major XA_sum / later B_fin
    float* B_fin = XA;                 // reuse (XA dead before FFN2)
    float* A_sum = XA + CN;            // N floats
    float* sc1   = A_sum + NN;
    float* sh1   = sc1 + 1024;
    float* sc2   = sh1 + 1024;
    float* sh2   = sc2 + 1024;
    float* E4    = B_t;                // 4C x N, overlays B_t..B_sig (all dead)

    dim3 blk256(256);
    dim3 blkConv(16, 16);
    dim3 grdConvC(WW / 16, HH / 16, CC);
    dim3 grdConv4C(WW / 16, HH / 16, 4 * CC);
    dim3 grdGemmC(NN / 64, CC / 32);
    dim3 grdGemmK(NN / 64, (KK + 31) / 32);
    dim3 blkTr(32, 8);
    dim3 grdEw(4096);

    // 0. row (N,C) -> planar (C,N)
    { dim3 g((CC + 31) / 32, (NN + 31) / 32);
      transpose_k<<<g, blkTr, 0, stream>>>(X, B_img, NN, CC); }

    // --- PPM ---
    dwconv17_k<<<grdConvC, blkConv, 0, stream>>>(B_img, ppm_w0, B_t);
    stats_planar_k<<<CC, blk256, 0, stream>>>(B_t, NN, ppm_g0, ppm_b0, sc1, sh1);
    gemm_wmma_k<1,1,0,0><<<grdGemmC, blk256, 0, stream>>>(ppm_w1, B_t, B_u, CC, CC,
                                                          sc1, sh1, nullptr);
    stats_planar_k<<<CC, blk256, 0, stream>>>(B_u, NN, ppm_g1, ppm_b1, sc2, sh2);
    add_bn_k<<<grdEw, blk256, 0, stream>>>(B_img, B_u, sc2, sh2);

    // --- FFN1 ---
    gemm_wmma_k<0,1,0,0><<<grdGemmC, blk256, 0, stream>>>(ffn1_w0, B_img, B_t, CC, CC,
                                                          nullptr, nullptr, nullptr);
    stats_planar_k<<<CC, blk256, 0, stream>>>(B_t, NN, ffn1_g0, ffn1_b0, sc1, sh1);
    gemm_wmma_k<1,1,0,0><<<grdGemmC, blk256, 0, stream>>>(ffn1_w1, B_t, B_u, CC, CC,
                                                          sc1, sh1, nullptr);
    stats_planar_k<<<CC, blk256, 0, stream>>>(B_u, NN, ffn1_g1, ffn1_b1, sc2, sh2);
    add_bn_k<<<grdEw, blk256, 0, stream>>>(B_img, B_u, sc2, sh2);

    // --- discriminator MLP ---
    gemm_wmma_k<0,0,1,1><<<grdGemmC, blk256, 0, stream>>>(dis_w1, B_img, B_h, CC, CC,
                                                          nullptr, nullptr, dis_b1);
    gemm_wmma_k<0,0,1,0><<<grdGemmK, blk256, 0, stream>>>(dis_w2, B_h, B_sig, KK, CC,
                                                          nullptr, nullptr, dis_b2);

    // --- edge gather/scatter (row-major for coalesced atomics) ---
    { dim3 g((NN + 31) / 32, (CC + 31) / 32);
      transpose_k<<<g, blkTr, 0, stream>>>(B_img, R_xn, CC, NN); }
    zero_k<<<grdEw, blk256, 0, stream>>>(XA, CN);
    zero_k<<<dim3(144), blk256, 0, stream>>>(A_sum, (size_t)NN);
    edge_k<<<dim3(EE / 8), blk256, 0, stream>>>(InfoIdx, edgmsk, B_sig, R_xn,
                                                XA, A_sum);
    stats_xtrans_k<<<CC, blk256, 0, stream>>>(XA, A_sum, bn_g, bn_b, sc1, sh1);
    xtrans_apply_k<<<grdEw, blk256, 0, stream>>>(R_xn, XA, A_sum, sc1, sh1, B_u);
    { dim3 g((CC + 31) / 32, (NN + 31) / 32);    // row (N,C) -> planar (C,N)
      transpose_k<<<g, blkTr, 0, stream>>>(B_u, B_img, NN, CC); }

    // --- FFN2 ---
    dwconv3x3_k<<<grdConv4C, blkConv, 0, stream>>>(B_img, ffn2_w0, E4);
    stats_planar_k<<<4 * CC, blk256, 0, stream>>>(E4, NN, ffn2_g0, ffn2_b0, sc1, sh1);
    gemm_wmma_k<1,1,0,0><<<grdGemmC, blk256, 0, stream>>>(ffn2_w1, E4, B_fin, CC, 4 * CC,
                                                          sc1, sh1, nullptr);
    stats_planar_k<<<CC, blk256, 0, stream>>>(B_fin, NN, ffn2_g1, ffn2_b1, sc2, sh2);

    // --- final BN apply + residual, planar -> row-major output ---
    { dim3 g(NN / 32, CC / 32);
      final_out_k<<<g, blkTr, 0, stream>>>(B_fin, B_img, sc2, sh2, out); }
}
